// MultiHeadAttention_31714038514323
// MI455X (gfx1250) — matrix operand order repaired and checked by reading, not compiled, not run on hardware
//
#include <hip/hip_runtime.h>
#include <hip/hip_bf16.h>

// ---------------------------------------------------------------------------
// MultiHeadAttention for MI455X (gfx1250): bf16 WMMA + TDM pipeline.
//   B=2, S=2048, D_IN=D_OUT=1024, H=16, HD=64
// ---------------------------------------------------------------------------

#define B_   2
#define S_   2048
#define DIN  1024
#define DOUT 1024
#define H_   16
#define HD_  64
#define NTOK (B_ * S_)
#define NP   (DIN / 32)          // k-panels per GEMM

typedef __attribute__((ext_vector_type(8)))  float        v8f;
typedef __attribute__((ext_vector_type(8)))  __bf16       v8bf;
typedef __attribute__((ext_vector_type(16))) __bf16       v16bf;
typedef __attribute__((ext_vector_type(4)))  unsigned int v4u;
typedef __attribute__((ext_vector_type(8)))  int          v8i;
typedef __attribute__((ext_vector_type(4)))  int          v4i;

static __device__ inline v16bf cat8(v8bf lo, v8bf hi) {
  return __builtin_shufflevector(lo, hi, 0,1,2,3,4,5,6,7,8,9,10,11,12,13,14,15);
}

static __device__ inline v8f wmma_bf16(v16bf a, v16bf b, v8f c) {
  // D = A(16x32) * B(32x16) + C(16x16), f32 accumulate
  return __builtin_amdgcn_wmma_f32_16x16x32_bf16(false, a, false, b,
                                                 (short)0, c, false, false);
}

// A-matrix 16x32 bf16 fragment from a row-major bf16 source.
// lanes 0-15: row=lane, k=0..15; lanes 16-31: row=lane-16, k=16..31
static __device__ inline v16bf afrag_bf16(const __bf16* rowp, int lane) {
  const int klo = (lane & 16) ? 16 : 0;
  v8bf c0 = *(const v8bf*)(rowp + klo);
  v8bf c1 = *(const v8bf*)(rowp + klo + 8);
  return cat8(c0, c1);
}

// Same A fragment converting from fp32 on the fly (v_cvt_pk_bf16_f32).
static __device__ inline v16bf afrag_f32(const float* rowp, int lane) {
  const int klo = (lane & 16) ? 16 : 0;
  v16bf r;
#pragma unroll
  for (int i = 0; i < 8; ++i) {
    r[i]     = (__bf16)rowp[klo + i];
    r[i + 8] = (__bf16)rowp[klo + 8 + i];
  }
  return r;
}

// B-matrix 32x16 bf16 fragment: lane holds column n=lane&15,
// k = 0..15 (lanes 0-15) or 16..31 (lanes 16-31) => 32 contiguous bytes.
static __device__ inline v16bf bfrag16(const __bf16* colp) {
  v8bf c0 = *(const v8bf*)(colp);
  v8bf c1 = *(const v8bf*)(colp + 8);
  return cat8(c0, c1);
}

// The TDM engine writes LDS behind the compiler's back (we hand it raw LDS
// byte offsets).  Without this, LLVM proves the staging buffer is never
// stored and folds the ds_loads to undef / hoists them out of the k-loop.
// Escaping the pointer into an asm with a memory clobber forces honest,
// per-iteration ds_load_b128s after each s_wait_tensorcnt + barrier.
static __device__ inline void lds_written_by_tdm(const __bf16* p) {
  asm volatile("" : : "v"((const void*)p) : "memory");
}

// ---------------------------------------------------------------------------
// TDM: async-copy one 64(n) x 32(k) bf16 tile of a column-major 1024x1024
// weight matrix into LDS at byte offset `lds_off`.  D# per ISA ch.8:
//   group0: count=1 | lds_addr | global_addr[56:0] | type=2
//   group1: data_size=2B, tensor_dim0/1=1024, tile=32x64, dim0_stride=1024
// Tracked by TENSORcnt; waited with s_wait_tensorcnt.
// ---------------------------------------------------------------------------
static __device__ inline void tdm_load_wtile(unsigned lds_off, const __bf16* gp) {
  const unsigned long long ga = (unsigned long long)(uintptr_t)gp;
  v4u g0;
  g0.x = 1u;                                       // count=1, user descriptor
  g0.y = lds_off;                                  // LDS byte address
  g0.z = (unsigned)ga;                             // global_addr[31:0]
  g0.w = (unsigned)((ga >> 32) & 0x01FFFFFFu) | (2u << 30);  // addr[56:32]|type=2
  v8i g1;
  g1[0] = (int)(1u << 16);                         // data_size=1 (2 bytes)
  g1[1] = (int)((1024u & 0xFFFFu) << 16);          // tensor_dim0[15:0] @bit48
  g1[2] = (int)((1024u >> 16) | ((1024u & 0xFFFFu) << 16)); // dim0_hi | dim1_lo
  g1[3] = (int)((1024u >> 16) | (32u << 16));      // dim1_hi | tile_dim0=32
  g1[4] = (int)(64u);                              // tile_dim1=64, tile_dim2=0
  g1[5] = (int)(1024u);                            // tensor_dim0_stride[31:0]
  g1[6] = 0;                                       // stride0_hi | stride1_lo
  g1[7] = 0;
  const v4i z4 = {0, 0, 0, 0};                     // 2D tensor: groups 2/3 unused
  const v8i z8 = {0, 0, 0, 0, 0, 0, 0, 0};
  __builtin_amdgcn_tensor_load_to_lds(g0, g1, z4, z4, z8, 0);
}

// ---------------------------------------------------------------------------
// Kernel 0: weight prep. Transpose + convert the four fp32 1024x1024 weight
// matrices to bf16 column-major Wt[sel][n][k] so GEMM B-panels are plain
// contiguous 2D tiles for the TDM.  Bandwidth-trivial (16 MB in, 8 MB out).
// ---------------------------------------------------------------------------
__global__ __launch_bounds__(256)
void wprep_kernel(const float* __restrict__ Wq, const float* __restrict__ Wk,
                  const float* __restrict__ Wv, const float* __restrict__ Wo,
                  __bf16* __restrict__ Wt) {
  __shared__ float tile[32][33];
  const int sel = blockIdx.z;
  const float* W = (sel == 0) ? Wq : (sel == 1) ? Wk : (sel == 2) ? Wv : Wo;
  __bf16* Wts = Wt + ((size_t)sel << 20);
  const int bx = blockIdx.x * 32, by = blockIdx.y * 32;
  const int tx = threadIdx.x & 31, ty = threadIdx.x >> 5;   // 32 x 8
#pragma unroll
  for (int i = 0; i < 32; i += 8)
    tile[ty + i][tx] = W[(size_t)(by + ty + i) * DOUT + bx + tx];
  __syncthreads();
#pragma unroll
  for (int i = 0; i < 32; i += 8)   // Wt[n][k] = W[k][n]; coalesced on k
    Wts[(size_t)(bx + ty + i) * DIN + by + tx] = (__bf16)tile[tx][ty + i];
}

// ---------------------------------------------------------------------------
// Kernel 1: fused QKV projection.  C[4096x1024] = x * W, blockIdx.z selects
// Wq/Wk/Wv.  128 threads = 4 waves; tile 128(M) x 64(N); each wave owns 32
// rows (2 A-frags, 8 accumulators -> 8 WMMAs/panel).  W panels arrive via
// double-buffered TDM loads issued by wave 0.  Q,K row-major bf16; V stored
// transposed per head Vt[b][h][d][s].
// ---------------------------------------------------------------------------
__global__ __launch_bounds__(128)
void qkv_proj_kernel(const float* __restrict__ x,
                     const __bf16* __restrict__ Wt,
                     __bf16* __restrict__ Qb,
                     __bf16* __restrict__ Kb,
                     __bf16* __restrict__ Vt) {
  __shared__ __bf16 Bt[2][64 * 32];        // double-buffered TDM tiles (8 KB)
  const int m0  = blockIdx.x * 128;
  const int n0  = blockIdx.y * 64;
  const int sel = blockIdx.z;
  const __bf16* Wn = Wt + ((size_t)sel << 20) + (size_t)n0 * DIN;

  const int tid = threadIdx.x;
  const int lane = tid & 31, wv = tid >> 5, ln = lane & 15;

  v8f acc[2][4] = {};

  if (wv == 0) tdm_load_wtile(0u, Wn);     // panel 0 -> buffer 0

  for (int t = 0; t < NP; ++t) {
    const int k0 = t * 32;
    if (wv == 0) {
      if (t + 1 < NP) {                    // issue next panel, wait current
        tdm_load_wtile(((t + 1) & 1) * (unsigned)(64 * 32 * 2),
                       Wn + (size_t)(t + 1) * 32);
        __builtin_amdgcn_s_wait_tensorcnt((short)1);
      } else {
        __builtin_amdgcn_s_wait_tensorcnt((short)0);
      }
    }
    __syncthreads();
    lds_written_by_tdm(&Bt[0][0]);         // TDM filled Bt[t&1]; force reload

    const float* ar0 = x + (size_t)(m0 + 32 * wv + ln) * DIN + k0;
    const float* ar1 = x + (size_t)(m0 + 32 * wv + 16 + ln) * DIN + k0;
    if (k0 + 32 < DIN) __builtin_prefetch(ar0 + 32, 0, 3);
    const v16bf a0 = afrag_f32(ar0, lane);
    const v16bf a1 = afrag_f32(ar1, lane);

    const __bf16* Bp = Bt[t & 1];
    const int kh = (lane & 16) ? 16 : 0;
    v16bf bf[4];
#pragma unroll
    for (int j = 0; j < 4; ++j)
      bf[j] = bfrag16(&Bp[(16 * j + ln) * 32 + kh]);
#pragma unroll
    for (int j = 0; j < 4; ++j) acc[0][j] = wmma_bf16(a0, bf[j], acc[0][j]);
#pragma unroll
    for (int j = 0; j < 4; ++j) acc[1][j] = wmma_bf16(a1, bf[j], acc[1][j]);
    __syncthreads();
  }

  const int hl = (lane & 16) ? 8 : 0;      // C-layout: row m = r + hl
#pragma unroll
  for (int i = 0; i < 2; ++i) {
#pragma unroll
    for (int j = 0; j < 4; ++j) {
#pragma unroll
      for (int r = 0; r < 8; ++r) {
        const int tok = m0 + 32 * wv + 16 * i + r + hl;
        const int col = n0 + 16 * j + ln;
        const __bf16 val = (__bf16)acc[i][j][r];
        if (sel == 0) {
          Qb[(size_t)tok * DOUT + col] = val;
        } else if (sel == 1) {
          Kb[(size_t)tok * DOUT + col] = val;
        } else {
          const int h = col >> 6, d = col & (HD_ - 1);
          const int b = tok >> 11, s = tok & (S_ - 1);
          Vt[(((size_t)(b * H_ + h) * HD_ + d) << 11) + s] = val;
        }
      }
    }
  }
}

// ---------------------------------------------------------------------------
// Kernel 2: flash attention. One wave per (b, h, 16-query tile).
// 32 keys/step: 4 WMMAs Q*K^T + online softmax (16-lane shuffle reductions)
// + 4 WMMAs P*V (P re-laid out via a 1 KB LDS tile; V pre-transposed).
// ---------------------------------------------------------------------------
__global__ __launch_bounds__(32)
void flash_attn_kernel(const __bf16* __restrict__ Qb,
                       const __bf16* __restrict__ Kb,
                       const __bf16* __restrict__ Vt,
                       const unsigned char* __restrict__ amask,
                       __bf16* __restrict__ Ctx) {
  __shared__ __bf16 Pt[16 * 32];

  const int lane = threadIdx.x & 31;
  const int ln   = lane & 15;
  const int hl   = (lane & 16) ? 8 : 0;
  const int q0   = blockIdx.x * 16;
  const int h    = blockIdx.y;
  const int b    = blockIdx.z;
  const int hc   = h * HD_;
  const int tq   = b * S_ + q0;

  v16bf qa[2];
#pragma unroll
  for (int kk = 0; kk < 2; ++kk)
    qa[kk] = afrag_bf16(Qb + (size_t)(tq + ln) * DOUT + hc + 32 * kk, lane);

  v8f   O[4] = {};
  float run_m[8], run_l[8];
#pragma unroll
  for (int r = 0; r < 8; ++r) { run_m[r] = -1e30f; run_l[r] = 0.0f; }

  const __bf16* Vh = Vt + ((size_t)(b * H_ + h) * HD_ << 11);
  const int nsteps = (q0 + 47) >> 5;       // ceil((q0+16)/32) causal steps

  for (int t = 0; t < nsteps; ++t) {
    const int kt0 = t * 32;

    v8f cA = {}, cB = {};
#pragma unroll
    for (int kk = 0; kk < 2; ++kk) {
      const int dlo = hc + 32 * kk + ((lane & 16) ? 16 : 0);
      const v16bf kb0 = bfrag16(Kb + (size_t)(b * S_ + kt0 + ln)      * DOUT + dlo);
      const v16bf kb1 = bfrag16(Kb + (size_t)(b * S_ + kt0 + 16 + ln) * DOUT + dlo);
      cA = wmma_bf16(qa[kk], kb0, cA);
      cB = wmma_bf16(qa[kk], kb1, cB);
    }

    const int keyA = kt0 + ln;
    const int keyB = kt0 + 16 + ln;
    const bool okA = amask[b * S_ + keyA] != 0;
    const bool okB = amask[b * S_ + keyB] != 0;
    const int qrow = q0 + hl;
#pragma unroll
    for (int r = 0; r < 8; ++r) {
      float sA = cA[r] * 0.125f;           // 1/sqrt(64)
      float sB = cB[r] * 0.125f;
      if (!okA || keyA > qrow + r) sA = -1e30f;
      if (!okB || keyB > qrow + r) sB = -1e30f;
      cA[r] = sA; cB[r] = sB;
    }

#pragma unroll
    for (int r = 0; r < 8; ++r) {
      float tm = fmaxf(cA[r], cB[r]);
      tm = fmaxf(tm, __shfl_xor(tm, 1));
      tm = fmaxf(tm, __shfl_xor(tm, 2));
      tm = fmaxf(tm, __shfl_xor(tm, 4));
      tm = fmaxf(tm, __shfl_xor(tm, 8));
      const float nm  = fmaxf(run_m[r], tm);
      const float fac = __expf(run_m[r] - nm);
      run_m[r] = nm;
      const float pA = __expf(cA[r] - nm);
      const float pB = __expf(cB[r] - nm);
      float rs = pA + pB;
      rs += __shfl_xor(rs, 1);
      rs += __shfl_xor(rs, 2);
      rs += __shfl_xor(rs, 4);
      rs += __shfl_xor(rs, 8);
      run_l[r] = run_l[r] * fac + rs;
#pragma unroll
      for (int j = 0; j < 4; ++j) O[j][r] *= fac;
      Pt[(r + hl) * 32 + ln]      = (__bf16)pA;
      Pt[(r + hl) * 32 + 16 + ln] = (__bf16)pB;
    }
    __syncthreads();

    const v16bf pa  = afrag_bf16(&Pt[ln * 32], lane);
    const int   klo = (lane & 16) ? 16 : 0;
#pragma unroll
    for (int j = 0; j < 4; ++j) {
      const v16bf vb = bfrag16(Vh + (size_t)(16 * j + ln) * S_ + kt0 + klo);
      O[j] = wmma_bf16(pa, vb, O[j]);
    }
    __syncthreads();
  }

#pragma unroll
  for (int r = 0; r < 8; ++r) {
    const float inv = 1.0f / run_l[r];
    const int   tok = tq + r + hl;
#pragma unroll
    for (int j = 0; j < 4; ++j)
      Ctx[(size_t)tok * DOUT + hc + 16 * j + ln] = (__bf16)(O[j][r] * inv);
  }
}

// ---------------------------------------------------------------------------
// Kernel 3: output projection. out = Ctx(bf16) * W_out + b_out, fp32 result.
// Same TDM-fed 128x64 tile structure as kernel 1.
// ---------------------------------------------------------------------------
__global__ __launch_bounds__(128)
void out_proj_kernel(const __bf16* __restrict__ Ctx,
                     const __bf16* __restrict__ Wto,   // transposed bf16 W_out
                     const float* __restrict__ bo,
                     float* __restrict__ out) {
  __shared__ __bf16 Bt[2][64 * 32];
  const int m0 = blockIdx.x * 128;
  const int n0 = blockIdx.y * 64;
  const __bf16* Wn = Wto + (size_t)n0 * DIN;

  const int tid = threadIdx.x;
  const int lane = tid & 31, wv = tid >> 5, ln = lane & 15;

  v8f acc[2][4] = {};

  if (wv == 0) tdm_load_wtile(0u, Wn);

  for (int t = 0; t < NP; ++t) {
    const int k0 = t * 32;
    if (wv == 0) {
      if (t + 1 < NP) {
        tdm_load_wtile(((t + 1) & 1) * (unsigned)(64 * 32 * 2),
                       Wn + (size_t)(t + 1) * 32);
        __builtin_amdgcn_s_wait_tensorcnt((short)1);
      } else {
        __builtin_amdgcn_s_wait_tensorcnt((short)0);
      }
    }
    __syncthreads();
    lds_written_by_tdm(&Bt[0][0]);

    const v16bf a0 = afrag_bf16(Ctx + (size_t)(m0 + 32 * wv + ln) * DOUT + k0, lane);
    const v16bf a1 = afrag_bf16(Ctx + (size_t)(m0 + 32 * wv + 16 + ln) * DOUT + k0, lane);

    const __bf16* Bp = Bt[t & 1];
    const int kh = (lane & 16) ? 16 : 0;
    v16bf bf[4];
#pragma unroll
    for (int j = 0; j < 4; ++j)
      bf[j] = bfrag16(&Bp[(16 * j + ln) * 32 + kh]);
#pragma unroll
    for (int j = 0; j < 4; ++j) acc[0][j] = wmma_bf16(a0, bf[j], acc[0][j]);
#pragma unroll
    for (int j = 0; j < 4; ++j) acc[1][j] = wmma_bf16(a1, bf[j], acc[1][j]);
    __syncthreads();
  }

  const int hl = (lane & 16) ? 8 : 0;
#pragma unroll
  for (int i = 0; i < 2; ++i) {
#pragma unroll
    for (int j = 0; j < 4; ++j) {
      const int   col  = n0 + 16 * j + ln;
      const float bias = bo[col];
#pragma unroll
      for (int r = 0; r < 8; ++r)
        out[(size_t)(m0 + 32 * wv + 16 * i + r + hl) * DOUT + col] =
            acc[i][j][r] + bias;
    }
  }
}

// ---------------------------------------------------------------------------
extern "C" void kernel_launch(void* const* d_in, const int* in_sizes, int n_in,
                              void* d_out, int out_size, void* d_ws, size_t ws_size,
                              hipStream_t stream) {
  const float*         x     = (const float*)d_in[0];
  const unsigned char* amask = (const unsigned char*)d_in[1];  // jnp bool = u8
  const float*         Wq    = (const float*)d_in[2];
  const float*         Wk    = (const float*)d_in[3];
  const float*         Wv    = (const float*)d_in[4];
  const float*         Wo    = (const float*)d_in[5];
  const float*         bo    = (const float*)d_in[6];
  float*               out   = (float*)d_out;

  // Workspace: Wt[4] (transposed bf16 weights, 2 MB each) + Q, K, V^T, Ctx
  // (bf16, 8 MB each) = 40 MB total.  Whole working set fits the 192 MB L2.
  __bf16* Wt  = (__bf16*)d_ws;                       // [4][1024][1024]
  __bf16* Qb  = Wt + ((size_t)4 << 20);
  __bf16* Kb  = Qb + (size_t)NTOK * DOUT;
  __bf16* Vt  = Kb + (size_t)NTOK * DOUT;
  __bf16* Ctx = Vt + (size_t)NTOK * DOUT;

  wprep_kernel<<<dim3(32, 32, 4), 256, 0, stream>>>(Wq, Wk, Wv, Wo, Wt);

  qkv_proj_kernel<<<dim3(NTOK / 128, DOUT / 64, 3), 128, 0, stream>>>(
      x, Wt, Qb, Kb, Vt);

  flash_attn_kernel<<<dim3(S_ / 16, H_, B_), 32, 0, stream>>>(
      Qb, Kb, Vt, amask, Ctx);

  out_proj_kernel<<<dim3(NTOK / 128, DOUT / 64, 1), 128, 0, stream>>>(
      Ctx, Wt + ((size_t)3 << 20), bo, out);
}
